// GAT_79937931313854
// MI455X (gfx1250) — compile-verified, matrix operand order checked by reading, not run on hardware
//
#include <hip/hip_runtime.h>
#include <hip/hip_bf16.h>

// ---------------- problem constants ----------------
#define N_NODES 50000
#define N_EDGES 800000
#define HEADS   4
#define FHID    32
#define CLS     40
#define IN_F    128
#define HF      128        // HEADS*FHID
#define HC      160        // HEADS*CLS
#define BN_EPS  1e-5f
#define SLOPE   0.2f

typedef __attribute__((ext_vector_type(16))) __bf16 v16bf;
typedef __attribute__((ext_vector_type(8)))  float  v8f;
typedef __attribute__((ext_vector_type(4)))  unsigned int u32x4;

union FragBF { v16bf v; u32x4 q[2]; };

// ordered-int encoding so atomicMax(u32) == float max (handles negatives)
__device__ __forceinline__ unsigned encf(float f) {
    unsigned u = __float_as_uint(f);
    return (u & 0x80000000u) ? ~u : (u | 0x80000000u);
}
__device__ __forceinline__ float decf(unsigned e) {
    unsigned u = (e & 0x80000000u) ? (e & 0x7FFFFFFFu) : ~e;
    return __uint_as_float(u);
}

// ---------------- conversion kernels ----------------
__global__ void k_f32_to_bf16(const float* __restrict__ x, __bf16* __restrict__ y, int n) {
    int t = blockIdx.x * blockDim.x + threadIdx.x;
    if (t < n) y[t] = (__bf16)x[t];
}

// W f32 [K x Nout] (row-major) -> Wt bf16 [Nout x K]
__global__ void k_w_transpose(const float* __restrict__ w, __bf16* __restrict__ wt,
                              int K, int Nout) {
    int t = blockIdx.x * blockDim.x + threadIdx.x;
    if (t >= K * Nout) return;
    int k = t / Nout, n = t % Nout;
    wt[n * K + k] = (__bf16)w[t];
}

// ---------------- fused dual WMMA GEMM ----------------
// Bt is [2*Nout x K] bf16: rows [0,Nout) = w_fc^T (-> Cft, no bias),
// rows [Nout,2*Nout) = w_lin^T (-> Crst, +bias).
// Each wave: 16 rows x 32 cols (one A fragment feeds two B fragments,
// two independent accumulators -> 8 WMMAs/wave, hides WMMA RAW latency).
__device__ __forceinline__ void store_tile(const v8f& acc, int col16, int l16, int kh,
                                           int rowBase, float* __restrict__ Cft,
                                           float* __restrict__ Crst,
                                           const float* __restrict__ bias, int Nout) {
    float* C;
    int col = col16 + l16;
    float badd;
    if (col16 < Nout) { C = Cft; badd = 0.0f; }
    else              { C = Crst; col -= Nout; badd = bias[col]; }
    float* p = C + (size_t)(rowBase + kh * 8) * Nout + col;
    #pragma unroll
    for (int r = 0; r < 8; ++r)
        p[(size_t)r * Nout] = acc[r] + badd;
}

__global__ __launch_bounds__(256) void k_gemm_dual(const __bf16* __restrict__ A,
                                                   const __bf16* __restrict__ Bt,
                                                   float* __restrict__ Cft,
                                                   float* __restrict__ Crst,
                                                   const float* __restrict__ bias,
                                                   int M, int K, int Nout) {
    const int lane = threadIdx.x & 31;
    const int wave = threadIdx.x >> 5;
    const int colBase = blockIdx.x * 32;               // 2 x 16-col subtiles
    const int rowBase = blockIdx.y * 128 + wave * 16;
    if (rowBase >= M) return;            // wave-uniform: EXEC all-ones for WMMA

    const int kh  = lane >> 4;           // half-wave: +8 K offset
    const int l16 = lane & 15;
    const __bf16* aRow  = A  + (size_t)(rowBase + l16) * K + kh * 8;
    const __bf16* bCol0 = Bt + (size_t)(colBase + l16) * K + kh * 8;
    const __bf16* bCol1 = bCol0 + (size_t)16 * K;

    v8f acc0 = {}, acc1 = {};
    #pragma unroll
    for (int k0 = 0; k0 < 128; k0 += 32) {
        FragBF a, b0, b1;
        a.q[0]  = *(const u32x4*)(aRow  + k0);
        a.q[1]  = *(const u32x4*)(aRow  + k0 + 16);
        b0.q[0] = *(const u32x4*)(bCol0 + k0);
        b0.q[1] = *(const u32x4*)(bCol0 + k0 + 16);
        b1.q[0] = *(const u32x4*)(bCol1 + k0);
        b1.q[1] = *(const u32x4*)(bCol1 + k0 + 16);
        acc0 = __builtin_amdgcn_wmma_f32_16x16x32_bf16(false, a.v, false, b0.v,
                                                       (short)0, acc0, false, false);
        acc1 = __builtin_amdgcn_wmma_f32_16x16x32_bf16(false, a.v, false, b1.v,
                                                       (short)0, acc1, false, false);
    }
    store_tile(acc0, colBase,      l16, kh, rowBase, Cft, Crst, bias, Nout);
    store_tile(acc1, colBase + 16, l16, kh, rowBase, Cft, Crst, bias, Nout);
}

// ---------------- attention projections: el/er [N,H] ----------------
__global__ void k_el_er(const float* __restrict__ ft, const float* __restrict__ al,
                        const float* __restrict__ ar, float* __restrict__ el,
                        float* __restrict__ er, int outf) {
    int t = blockIdx.x * blockDim.x + threadIdx.x;
    if (t >= N_NODES * HEADS) return;
    int n = t / HEADS, h = t % HEADS;
    const float* f = ft + (size_t)n * HEADS * outf + h * outf;
    const float* a = al + h * outf;
    const float* b = ar + h * outf;
    float sl = 0.f, sr = 0.f;
    for (int i = 0; i < outf; ++i) { float v = f[i]; sl += v * a[i]; sr += v * b[i]; }
    el[t] = sl; er[t] = sr;
}

// ---------------- segment softmax ----------------
__global__ void k_init_seg(unsigned* __restrict__ mEnc, float* __restrict__ ssum) {
    int t = blockIdx.x * blockDim.x + threadIdx.x;
    if (t >= N_NODES * HEADS) return;
    mEnc[t] = 0x007FFFFFu;   // encf(-inf)
    ssum[t] = 0.0f;
}

__global__ void k_edge_max(const int* __restrict__ src, const int* __restrict__ dst,
                           const float* __restrict__ el, const float* __restrict__ er,
                           float* __restrict__ sbuf, unsigned* __restrict__ mEnc) {
    int e = blockIdx.x * blockDim.x + threadIdx.x;
    if (e >= N_EDGES) return;
    int s = src[e], d = dst[e];
    #pragma unroll
    for (int h = 0; h < HEADS; ++h) {
        float v = el[s * HEADS + h] + er[d * HEADS + h];
        v = v > 0.f ? v : SLOPE * v;
        sbuf[e * HEADS + h] = v;
        atomicMax(&mEnc[d * HEADS + h], encf(v));
    }
}

__global__ void k_edge_exp(const int* __restrict__ dst, float* __restrict__ sbuf,
                           const unsigned* __restrict__ mEnc, float* __restrict__ ssum) {
    int t = blockIdx.x * blockDim.x + threadIdx.x;
    if (t >= N_EDGES * HEADS) return;
    int e = t >> 2, h = t & 3;
    int d = dst[e];
    float m = decf(mEnc[d * HEADS + h]);
    float ex = expf(sbuf[t] - m);
    sbuf[t] = ex;                               // in-place: s -> exp(s-m)
    atomicAdd(&ssum[d * HEADS + h], ex);
}

// one thread = one (edge, 4-feature chunk); float4 read + 4 L2 float atomics
__global__ void k_edge_scatter(const int* __restrict__ src, const int* __restrict__ dst,
                               const float* __restrict__ sbuf, const float* __restrict__ ssum,
                               const float* __restrict__ ft, float* __restrict__ rst,
                               int outf) {
    const int cnt = (HEADS * outf) >> 2;
    long long t = (long long)blockIdx.x * blockDim.x + threadIdx.x;
    if (t >= (long long)N_EDGES * cnt) return;
    int e  = (int)(t / cnt);
    int c4 = (int)(t % cnt);
    int h  = (c4 * 4) / outf;                   // outf % 4 == 0, chunk never crosses heads
    int s = src[e], d = dst[e];
    float a = sbuf[e * HEADS + h] / ssum[d * HEADS + h];
    float4 f = *(const float4*)(ft + (size_t)s * HEADS * outf + c4 * 4);
    float* rp = rst + (size_t)d * HEADS * outf + c4 * 4;
    atomicAdd(rp + 0, f.x * a);
    atomicAdd(rp + 1, f.y * a);
    atomicAdd(rp + 2, f.z * a);
    atomicAdd(rp + 3, f.w * a);
}

// ---------------- batchnorm ----------------
__global__ __launch_bounds__(256) void k_bn_stats(const float* __restrict__ x,
                                                  float* __restrict__ mu,
                                                  float* __restrict__ istd) {
    __shared__ float s1[256], s2[256];
    const int col = blockIdx.x;
    float a = 0.f, b = 0.f;
    for (int n = threadIdx.x; n < N_NODES; n += 256) {
        float v = x[(size_t)n * HF + col];
        a += v; b += v * v;
    }
    s1[threadIdx.x] = a; s2[threadIdx.x] = b;
    __syncthreads();
    for (int st = 128; st > 0; st >>= 1) {
        if (threadIdx.x < st) {
            s1[threadIdx.x] += s1[threadIdx.x + st];
            s2[threadIdx.x] += s2[threadIdx.x + st];
        }
        __syncthreads();
    }
    if (threadIdx.x == 0) {
        float m = s1[0] / (float)N_NODES;
        float v = s2[0] / (float)N_NODES - m * m;
        mu[col] = m;
        istd[col] = rsqrtf(v + BN_EPS);
    }
}

__global__ void k_bn_apply(const float* __restrict__ x, const float* __restrict__ mu,
                           const float* __restrict__ istd, const float* __restrict__ g,
                           const float* __restrict__ be, __bf16* __restrict__ out) {
    int t = blockIdx.x * blockDim.x + threadIdx.x;
    if (t >= N_NODES * HF) return;
    int col = t % HF;
    float y = (x[t] - mu[col]) * istd[col] * g[col] + be[col];
    out[t] = (__bf16)(y > 0.f ? y : 0.f);
}

// ---------------- final head-mean + log_softmax ----------------
__global__ void k_final(const float* __restrict__ rst, const float* __restrict__ bias_last,
                        float* __restrict__ out) {
    int n = blockIdx.x * blockDim.x + threadIdx.x;
    if (n >= N_NODES) return;
    const float* r = rst + (size_t)n * HC;
    float v[CLS];
    #pragma unroll
    for (int c = 0; c < CLS; ++c)
        v[c] = 0.25f * (r[c] + r[CLS + c] + r[2 * CLS + c] + r[3 * CLS + c]) + bias_last[c];
    float m = v[0];
    #pragma unroll
    for (int c = 1; c < CLS; ++c) m = fmaxf(m, v[c]);
    float se = 0.f;
    #pragma unroll
    for (int c = 0; c < CLS; ++c) se += expf(v[c] - m);
    float lse = m + logf(se);
    #pragma unroll
    for (int c = 0; c < CLS; ++c) out[(size_t)n * CLS + c] = v[c] - lse;
}

// ---------------- workspace layout (bytes) ----------------
// NOTE: each layer's fc^T / lin^T weight blocks are contiguous so the fused
// dual-GEMM can index them as a single [2*Nout x K] matrix.
static constexpr size_t HBF_OFF = 0;                                        // N*128 bf16
static constexpr size_t WT0_OFF = HBF_OFF + (size_t)N_NODES * IN_F * 2;     // fc0^T
static constexpr size_t WT1_OFF = WT0_OFF + 128 * 128 * 2;                  // lin0^T
static constexpr size_t WT2_OFF = WT1_OFF + 128 * 128 * 2;                  // fc1^T
static constexpr size_t WT3_OFF = WT2_OFF + 128 * 128 * 2;                  // lin1^T
static constexpr size_t WT4_OFF = WT3_OFF + 128 * 128 * 2;                  // fc2^T (128x160)
static constexpr size_t WT5_OFF = WT4_OFF + 128 * 160 * 2;                  // lin2^T
static constexpr size_t FT_OFF  = WT5_OFF + 128 * 160 * 2;                  // N*160 f32
static constexpr size_t RST_OFF = FT_OFF + (size_t)N_NODES * HC * 4;        // N*160 f32
static constexpr size_t EL_OFF  = RST_OFF + (size_t)N_NODES * HC * 4;       // N*H f32
static constexpr size_t ER_OFF  = EL_OFF + (size_t)N_NODES * HEADS * 4;
static constexpr size_t ME_OFF  = ER_OFF + (size_t)N_NODES * HEADS * 4;     // u32
static constexpr size_t SS_OFF  = ME_OFF + (size_t)N_NODES * HEADS * 4;     // f32
static constexpr size_t SB_OFF  = SS_OFF + (size_t)N_NODES * HEADS * 4;     // E*H f32
static constexpr size_t MU_OFF  = SB_OFF + (size_t)N_EDGES * HEADS * 4;
static constexpr size_t IS_OFF  = MU_OFF + 256 * 4;

static inline int cdiv(long long a, int b) { return (int)((a + b - 1) / b); }

static void run_gat_layer(hipStream_t stream, const __bf16* hbf,
                          const __bf16* wt_pair,   // [2*Nout x K] bf16 (fc^T then lin^T)
                          const float* al, const float* ar, const float* b, int outf,
                          const int* src, const int* dst,
                          float* ft, float* rst, float* el, float* er,
                          unsigned* mEnc, float* ssum, float* sbuf) {
    const int Nout = HEADS * outf;
    dim3 gg((2 * Nout) / 32, (N_NODES + 127) / 128);
    k_gemm_dual<<<gg, 256, 0, stream>>>(hbf, wt_pair, ft, rst, b, N_NODES, 128, Nout);
    const int nh = N_NODES * HEADS;
    k_el_er<<<cdiv(nh, 256), 256, 0, stream>>>(ft, al, ar, el, er, outf);
    k_init_seg<<<cdiv(nh, 256), 256, 0, stream>>>(mEnc, ssum);
    k_edge_max<<<cdiv(N_EDGES, 256), 256, 0, stream>>>(src, dst, el, er, sbuf, mEnc);
    k_edge_exp<<<cdiv((long long)N_EDGES * HEADS, 256), 256, 0, stream>>>(dst, sbuf, mEnc, ssum);
    const int cnt = (HEADS * outf) / 4;
    k_edge_scatter<<<cdiv((long long)N_EDGES * cnt, 256), 256, 0, stream>>>(
        src, dst, sbuf, ssum, ft, rst, outf);
}

extern "C" void kernel_launch(void* const* d_in, const int* in_sizes, int n_in,
                              void* d_out, int out_size, void* d_ws, size_t ws_size,
                              hipStream_t stream) {
    const float* feat      = (const float*)d_in[0];
    const int*   src       = (const int*)d_in[1];
    const int*   dst       = (const int*)d_in[2];
    const float* w_fc0     = (const float*)d_in[3];
    const float* al0       = (const float*)d_in[4];
    const float* ar0       = (const float*)d_in[5];
    const float* b0        = (const float*)d_in[6];
    const float* w_lin0    = (const float*)d_in[7];
    const float* g0        = (const float*)d_in[8];
    const float* be0       = (const float*)d_in[9];
    const float* w_fc1     = (const float*)d_in[10];
    const float* al1       = (const float*)d_in[11];
    const float* ar1       = (const float*)d_in[12];
    const float* b1        = (const float*)d_in[13];
    const float* w_lin1    = (const float*)d_in[14];
    const float* g1        = (const float*)d_in[15];
    const float* be1       = (const float*)d_in[16];
    const float* w_fc2     = (const float*)d_in[17];
    const float* al2       = (const float*)d_in[18];
    const float* ar2       = (const float*)d_in[19];
    const float* b2        = (const float*)d_in[20];
    const float* w_lin2    = (const float*)d_in[21];
    const float* bias_last = (const float*)d_in[22];
    float* out = (float*)d_out;

    char* ws = (char*)d_ws;
    __bf16*   hbf   = (__bf16*)(ws + HBF_OFF);
    __bf16*   wt0   = (__bf16*)(ws + WT0_OFF);
    __bf16*   wt1   = (__bf16*)(ws + WT1_OFF);
    __bf16*   wt2   = (__bf16*)(ws + WT2_OFF);
    __bf16*   wt3   = (__bf16*)(ws + WT3_OFF);
    __bf16*   wt4   = (__bf16*)(ws + WT4_OFF);
    __bf16*   wt5   = (__bf16*)(ws + WT5_OFF);
    float*    ft    = (float*)(ws + FT_OFF);
    float*    rst   = (float*)(ws + RST_OFF);
    float*    el    = (float*)(ws + EL_OFF);
    float*    er    = (float*)(ws + ER_OFF);
    unsigned* mEnc  = (unsigned*)(ws + ME_OFF);
    float*    ssum  = (float*)(ws + SS_OFF);
    float*    sbuf  = (float*)(ws + SB_OFF);
    float*    mu    = (float*)(ws + MU_OFF);
    float*    istd  = (float*)(ws + IS_OFF);

    // -- input conversion / weight pre-transpose (bf16) --
    k_f32_to_bf16<<<cdiv(N_NODES * IN_F, 256), 256, 0, stream>>>(feat, hbf, N_NODES * IN_F);
    k_w_transpose<<<cdiv(128 * 128, 256), 256, 0, stream>>>(w_fc0,  wt0, 128, 128);
    k_w_transpose<<<cdiv(128 * 128, 256), 256, 0, stream>>>(w_lin0, wt1, 128, 128);
    k_w_transpose<<<cdiv(128 * 128, 256), 256, 0, stream>>>(w_fc1,  wt2, 128, 128);
    k_w_transpose<<<cdiv(128 * 128, 256), 256, 0, stream>>>(w_lin1, wt3, 128, 128);
    k_w_transpose<<<cdiv(128 * 160, 256), 256, 0, stream>>>(w_fc2,  wt4, 128, 160);
    k_w_transpose<<<cdiv(128 * 160, 256), 256, 0, stream>>>(w_lin2, wt5, 128, 160);

    // -- layer 0 --
    run_gat_layer(stream, hbf, wt0, al0, ar0, b0, FHID, src, dst,
                  ft, rst, el, er, mEnc, ssum, sbuf);
    k_bn_stats<<<HF, 256, 0, stream>>>(rst, mu, istd);
    k_bn_apply<<<cdiv(N_NODES * HF, 256), 256, 0, stream>>>(rst, mu, istd, g0, be0, hbf);

    // -- layer 1 --
    run_gat_layer(stream, hbf, wt2, al1, ar1, b1, FHID, src, dst,
                  ft, rst, el, er, mEnc, ssum, sbuf);
    k_bn_stats<<<HF, 256, 0, stream>>>(rst, mu, istd);
    k_bn_apply<<<cdiv(N_NODES * HF, 256), 256, 0, stream>>>(rst, mu, istd, g1, be1, hbf);

    // -- layer 2 + readout --
    run_gat_layer(stream, hbf, wt4, al2, ar2, b2, CLS, src, dst,
                  ft, rst, el, er, mEnc, ssum, sbuf);
    k_final<<<cdiv(N_NODES, 256), 256, 0, stream>>>(rst, bias_last, out);
}
